// IntSoftmax_IBERT_10325101379983
// MI455X (gfx1250) — compile-verified, matrix operand order checked by reading, not run on hardware
//
#include <hip/hip_runtime.h>
#include <cmath>

// IntSoftmax (I-BERT) for x:(8,12,1024,1024) f32, sf scalar.
// One wave32 per 1024-element row; NT b128 streaming; row-sum via
// v_wmma_f32_16x16x32_f16 (exact: summands are small integers).
// All per-element divisions strength-reduced to multiplies by
// wave-scalar reciprocals (divisors are row-invariant) to stay under
// the 23.3 TB/s HBM roofline instead of going VALU-bound.

typedef __attribute__((ext_vector_type(4)))  float    v4f;
typedef __attribute__((ext_vector_type(8)))  float    v8f;
typedef __attribute__((ext_vector_type(16))) _Float16 v16h;

#define ROW      1024
#define WAVES_PB 8          // 256 threads = 8 wave32 rows per block

__global__ __launch_bounds__(256) void
isoftmax_ibert_kernel(const float* __restrict__ x,
                      const float* __restrict__ sfp,
                      float* __restrict__ out,
                      int nrows, size_t out_scalar_idx)
{
    const int lane = threadIdx.x & 31;
    const int wave = threadIdx.x >> 5;
    const int row  = blockIdx.x * WAVES_PB + wave;

    if (blockIdx.x == 0 && threadIdx.x == 0)
        out[out_scalar_idx] = 0.0078125f;          // out_scale = 2/256

    if (row >= nrows) return;                      // wave-uniform: EXEC stays all-1s

    // ---- scalar constants (few IEEE divs per thread; wave-uniform) ----
    const float sf        = *sfp;
    const float ACT_SCALE = (float)(756987985920.0 / 127.0);
    const float inv_sf    = 1.0f / sf;
    const float x0_int    = floorf(-0.6931f / sf);                         // negative
    const float b_int     = floorf((float)(0.96963238 / 0.35815147) / sf);
    const float c_int     = floorf((float)(1.0 / 0.35815147) / (sf * sf));
    const float exp_scale = (0.35815147f * sf * sf) / 1073741824.0f;       // /2^30
    const float new_scale = exp_scale / ACT_SCALE;
    const float t_req     = new_scale / exp_scale;  // requant multiplier (~1/ACT_SCALE)
    const float clamp_lo  = 30.0f * x0_int;
    // Nudged reciprocal: guarantees exact floor-division of the small
    // integer quotient xi/x0_int (overshoot << 1/|x0_int|).
    const float inv_x0_adj = (1.0f / x0_int) * (1.0f + 4.0e-7f);
    const float c_out      = 127.0f / ACT_SCALE;    // exp_q -> exp_int scale

    const v4f* __restrict__ xr  = (const v4f*)(x + (size_t)row * ROW);
    v4f* __restrict__       orr = (v4f*)(out + (size_t)row * ROW);

    // ---- pass 1: stream row (NT, b128), x_int = trunc(x/sf), row max ----
    v4f d[8];
#pragma unroll
    for (int c = 0; c < 8; ++c)
        d[c] = __builtin_nontemporal_load(&xr[c * 32 + lane]);

    float mx = -3.402823466e38f;
#pragma unroll
    for (int c = 0; c < 8; ++c) {
#pragma unroll
        for (int j = 0; j < 4; ++j) {
            float xi = truncf(d[c][j] * inv_sf);
            d[c][j]  = xi;
            mx = fmaxf(mx, xi);
        }
    }
#pragma unroll
    for (int off = 16; off >= 1; off >>= 1)
        mx = fmaxf(mx, __shfl_xor(mx, off, 32));

    // ---- pass 2: int-exp + QuantAct(8) -> small integer k per element ----
    v16h a0, a1;
#pragma unroll
    for (int c = 0; c < 8; ++c) {
#pragma unroll
        for (int j = 0; j < 4; ++j) {
            const int idx = c * 4 + j;
            float xi = d[c][j] - mx;                       // <= 0
            xi = fmaxf(xi, clamp_lo);
            float q = floorf(xi * inv_x0_adj);             // exact floor-div, in [0,30]
            float r = xi - x0_int * q;                     // exact small int
            float z = r * (r + b_int) + c_int;
            float e = fmaxf(floorf(ldexpf(z, 30 - (int)q)), 0.0f); // exact 2^(30-q) scale
            float k = rintf(e * t_req);                    // jnp.round == RNE
            d[c][j] = k;                                   // keep for output pass
            if (idx < 16) a0[idx]      = (_Float16)k;      // exact: small int
            else          a1[idx - 16] = (_Float16)k;
        }
    }

    // ---- row sum of k via WMMA (exact integer sums in f32 accumulators) ----
    v16h ones;
#pragma unroll
    for (int i = 0; i < 16; ++i) ones[i] = (_Float16)1.0f;

    v8f acc = {0.f, 0.f, 0.f, 0.f, 0.f, 0.f, 0.f, 0.f};
    // D[m][n] = rowsum_m(A); chained C accumulates both 512-element halves.
    acc = __builtin_amdgcn_wmma_f32_16x16x32_f16(false, a0, false, ones,
                                                 (short)0, acc, false, false);
    acc = __builtin_amdgcn_wmma_f32_16x16x32_f16(false, a1, false, ones,
                                                 (short)0, acc, false, false);
    float s = acc[0] + acc[1] + acc[2] + acc[3] +
              acc[4] + acc[5] + acc[6] + acc[7];           // M=0..7 (or 8..15) of one column
    s += __shfl_xor(s, 16, 32);                            // full column = total of all 1024 k

    const float exp_sum = (s * 127.0f) / ACT_SCALE;        // scalar, once
    const float factor  = floorf(4294967296.0f / exp_sum); // floor(2^32 / sum), scalar div
    const float fm      = factor * 0x1p-25f;               // /2^25, exact exponent shift

    // ---- pass 3: renormalize + NT b128 store (4 VALU ops/element) ----
#pragma unroll
    for (int c = 0; c < 8; ++c) {
        v4f o;
#pragma unroll
        for (int j = 0; j < 4; ++j) {
            float ei = d[c][j] * c_out;                    // requantized exp_int
            o[j] = floorf(ei * fm) * 0.0078125f;           // * out_scale (2/256)
        }
        __builtin_nontemporal_store(o, &orr[c * 32 + lane]);
    }
}

extern "C" void kernel_launch(void* const* d_in, const int* in_sizes, int n_in,
                              void* d_out, int out_size, void* d_ws, size_t ws_size,
                              hipStream_t stream)
{
    const float* x  = (const float*)d_in[0];
    const float* sf = (const float*)d_in[1];
    float* out      = (float*)d_out;

    const int nrows = in_sizes[0] / ROW;                   // 8*12*1024 = 98304
    const size_t out_scalar_idx = (size_t)nrows * ROW;     // trailing out_scale slot
    const int blocks = (nrows + WAVES_PB - 1) / WAVES_PB;  // 12288

    isoftmax_ibert_kernel<<<blocks, 256, 0, stream>>>(x, sf, out, nrows,
                                                      out_scalar_idx);
}
